// DynamicChunker_63788854280485
// MI455X (gfx1250) — matrix operand order repaired and checked by reading, not compiled, not run on hardware
//
#include <hip/hip_runtime.h>
#include <cstddef>

// ---------------- problem constants (match reference) ----------------
#define BB 2
#define TT 128
#define DD 256
#define LL 8
#define HH 4
#define DHH 64
#define FF 2048
#define SS 9          // 1 + L
#define THRESH 0.5f

typedef __attribute__((ext_vector_type(16))) _Float16 v16h;
typedef __attribute__((ext_vector_type(8)))  _Float16 v8h;
typedef __attribute__((ext_vector_type(8)))  float    v8f;

// ---------------- shared-memory layout (dynamic LDS, ~173 KB) ----------------
// All WMMA A-operand buffers are padded to 16 rows; pad rows zeroed once.
struct Smem {
    float    seq[SS * DD];      // [CLS; window] rows
    float    h[16 * DD];        // ln1(seq), rows 9..15 zero
    float    Km[SS * DD];       // K projection
    float    Vm[SS * DD];       // V projection
    float    Q0[DD];            // Q row 0 (CLS)
    float    sc[HH * SS];       // raw scores per head/key
    float    attw[LL * HH * SS];// softmax weights per candidate/head/key
    float    O[16 * DD];        // attention out per candidate, rows 8..15 zero
    float    X0[LL * DD];       // seq[0] + O@Wo + bo
    float    H2[16 * DD];       // ln2(X0), rows 8..15 zero
    _Float16 Y[16 * FF];        // relu(H2@W1+b1) f16, rows 8..15 zero
    float    ENC[16 * DD];      // X0 + Y@W2 + b2, rows 8..15 zero
    float    U[LL * DD];        // relu(ENC@We1+be1)
    float    p[LL];             // eoc probabilities
    int      ctrl[4];           // [i, count, lstar, active]
};

// ---------------- weight transpose+convert prologue ----------------
// dst[n*K + k] = (f16) src[k*N + n]   (B-fragments contiguous in K)
__global__ void dc_transpose_f16(const float* __restrict__ src,
                                 _Float16* __restrict__ dst, int K, int N) {
    int idx = blockIdx.x * blockDim.x + threadIdx.x;
    if (idx >= K * N) return;
    int k = idx / N, n = idx - k * N;
    dst[(size_t)n * K + k] = (_Float16)src[idx];
}

// ---------------- WMMA fragment helpers (CDNA5 16x16x32 f16 layouts) ----------
// A (16x32, 16-bit): lane<16 -> M=lane, halves 0..7 = K(k0+hi*8..), halves 8..15 = +16
__device__ __forceinline__ v16h frag_from_f32(const float* p0) {
    float4 x0 = *(const float4*)(p0);
    float4 x1 = *(const float4*)(p0 + 4);
    float4 x2 = *(const float4*)(p0 + 16);
    float4 x3 = *(const float4*)(p0 + 20);
    v16h a;
    a[0]  = (_Float16)x0.x; a[1]  = (_Float16)x0.y;
    a[2]  = (_Float16)x0.z; a[3]  = (_Float16)x0.w;
    a[4]  = (_Float16)x1.x; a[5]  = (_Float16)x1.y;
    a[6]  = (_Float16)x1.z; a[7]  = (_Float16)x1.w;
    a[8]  = (_Float16)x2.x; a[9]  = (_Float16)x2.y;
    a[10] = (_Float16)x2.z; a[11] = (_Float16)x2.w;
    a[12] = (_Float16)x3.x; a[13] = (_Float16)x3.y;
    a[14] = (_Float16)x3.z; a[15] = (_Float16)x3.w;
    return a;
}

// Preload the full K=256 A-operand (8 fragments, 64 VGPRs) once per GEMM phase.
__device__ __forceinline__ void preload_a8_f32(const float* A, int ld, int lane,
                                               v16h af[8]) {
    int m = lane & 15, hi = lane >> 4;
    const float* base = A + m * ld + (hi << 3);
#pragma unroll
    for (int kk = 0; kk < 8; ++kk) af[kk] = frag_from_f32(base + kk * 32);
}

// Branch-free f16 A-fragment load from padded LDS buffer.
__device__ __forceinline__ v16h load_a_f16p(const _Float16* A, int ld, int k0,
                                            int lane) {
    int m = lane & 15, hi = lane >> 4;
    const _Float16* p0 = A + m * ld + k0 + (hi << 3);
    v8h lo = *(const v8h*)p0;
    v8h hh = *(const v8h*)(p0 + 16);
    v16h a;
#pragma unroll
    for (int j = 0; j < 8; ++j) { a[j] = lo[j]; a[8 + j] = hh[j]; }
    return a;
}

// B (32x16, 16-bit): lane<16 -> N=lane, halves 0..7 = K(k0+hi*8..), halves 8..15 = +16.
// Weights pre-transposed [N][K] f16 -> two aligned 16B global loads per lane.
__device__ __forceinline__ v16h load_b_tr(const _Float16* Wt, int Kdim,
                                          int n0, int k0, int lane) {
    int n = lane & 15, hi = lane >> 4;
    const _Float16* p0 = Wt + (size_t)(n0 + n) * Kdim + k0 + (hi << 3);
    v8h lo = *(const v8h*)p0;
    v8h hh = *(const v8h*)(p0 + 16);
    v16h b;
#pragma unroll
    for (int j = 0; j < 8; ++j) { b[j] = lo[j]; b[8 + j] = hh[j]; }
    return b;
}

__device__ __forceinline__ v8f wmma_step(v16h a, v16h b, v8f c) {
    return __builtin_amdgcn_wmma_f32_16x16x32_f16(false, a, false, b,
                                                  (short)0, c, false, false);
}

// ---------------- wave-level layernorm over one 256-wide row ----------------
__device__ __forceinline__ void ln_row(const float* x, float* y,
                                       const float* g, const float* bt, int lane) {
    float s = 0.f, s2 = 0.f;
#pragma unroll
    for (int q = 0; q < 8; ++q) {
        float v = x[lane + 32 * q];
        s += v; s2 += v * v;
    }
#pragma unroll
    for (int off = 16; off; off >>= 1) {
        s  += __shfl_xor(s,  off, 32);
        s2 += __shfl_xor(s2, off, 32);
    }
    float mean = s * (1.f / 256.f);
    float var  = s2 * (1.f / 256.f) - mean * mean;
    float rs   = rsqrtf(var + 1e-5f);
#pragma unroll
    for (int q = 0; q < 8; ++q) {
        int e = lane + 32 * q;
        y[e] = (x[e] - mean) * rs * g[e] + bt[e];
    }
}

// ---------------- main kernel: one block == one batch chain ----------------
__global__ __launch_bounds__(256)
void dc_chunker(const float* __restrict__ frames, const float* __restrict__ cls,
                const float* __restrict__ bq, const float* __restrict__ bk,
                const float* __restrict__ bv, const float* __restrict__ bo,
                const float* __restrict__ ln1_g, const float* __restrict__ ln1_b,
                const float* __restrict__ ln2_g, const float* __restrict__ ln2_b,
                const float* __restrict__ b1f, const float* __restrict__ b2f,
                const float* __restrict__ be1, const float* __restrict__ We2,
                const float* __restrict__ be2,
                const _Float16* __restrict__ wqT, const _Float16* __restrict__ wkT,
                const _Float16* __restrict__ wvT, const _Float16* __restrict__ woT,
                const _Float16* __restrict__ w1T, const _Float16* __restrict__ w2T,
                const _Float16* __restrict__ we1T,
                float* __restrict__ out) {
    extern __shared__ char smem_raw[];
    Smem& S = *reinterpret_cast<Smem*>(smem_raw);

    const int b    = blockIdx.x;
    const int tid  = threadIdx.x;
    const int w    = tid >> 5;   // wave id 0..7
    const int lane = tid & 31;
    const int nn   = lane & 15;  // C/D & B column within tile
    const int hi   = lane >> 4;

    // ---- one-time init: ctrl + zero the A-operand padding rows ----
    if (tid == 0) { S.ctrl[0] = 0; S.ctrl[1] = 0; }
    for (int idx = tid; idx < 7 * DD; idx += 256) S.h[SS * DD + idx] = 0.f;
    for (int idx = tid; idx < 8 * DD; idx += 256) {
        S.O[LL * DD + idx]   = 0.f;
        S.H2[LL * DD + idx]  = 0.f;
        S.ENC[LL * DD + idx] = 0.f;
    }
    for (int idx = tid; idx < 8 * FF; idx += 256) S.Y[LL * FF + idx] = (_Float16)0.f;
    __syncthreads();

    for (int step = 0; step < TT; ++step) {
        // ---- 1. build seq = [CLS; frames[i:i+L] (zero past T)] ----
        {
            int i = S.ctrl[0];
            for (int idx = tid; idx < SS * DD; idx += 256) {
                int row = idx >> 8, col = idx & 255;
                float v;
                if (row == 0) v = cls[col];
                else {
                    int f = i + row - 1;
                    v = (f < TT) ? frames[((size_t)b * TT + f) * DD + col] : 0.f;
                }
                S.seq[idx] = v;
            }
        }
        __syncthreads();

        // ---- 2. LN1: h = ln(seq) ----
        for (int row = w; row < SS; row += 8)
            ln_row(S.seq + row * DD, S.h + row * DD, ln1_g, ln1_b, lane);
        __syncthreads();

        // ---- 3. QKV GEMMs (M=9->16, N=256, K=256): shared A, 48 tiles ----
        {
            v16h af[8];
            preload_a8_f32(S.h, DD, lane, af);
            for (int tile = w; tile < 48; tile += 8) {
                int which = tile >> 4;              // 0=Q, 1=K, 2=V
                int n0 = (tile & 15) << 4;
                const _Float16* Wt = (which == 0) ? wqT : (which == 1) ? wkT : wvT;
                v8f acc = {};
#pragma unroll
                for (int kk = 0; kk < 8; ++kk)
                    acc = wmma_step(af[kk], load_b_tr(Wt, DD, n0, kk * 32, lane), acc);
#pragma unroll
                for (int r = 0; r < 8; ++r) {
                    int row = r + (hi << 3), col = n0 + nn;
                    if (which == 0) {
                        if (row == 0) S.Q0[col] = acc[r] + bq[col];
                    } else if (which == 1) {
                        if (row < SS) S.Km[row * DD + col] = acc[r] + bk[col];
                    } else {
                        if (row < SS) S.Vm[row * DD + col] = acc[r] + bv[col];
                    }
                }
            }
        }
        __syncthreads();

        // ---- 4a. scores for CLS query: 4 heads x 9 keys ----
        if (tid < HH * SS) {
            int hh = tid / SS, j = tid - hh * SS;
            const float* q  = S.Q0 + hh * DHH;
            const float* kk = S.Km + j * DD + hh * DHH;
            float s = 0.f;
            for (int e = 0; e < DHH; ++e) s += q[e] * kk[e];
            S.sc[hh * SS + j] = s * 0.125f;  // 1/sqrt(64)
        }
        __syncthreads();

        // ---- 4b. masked softmax per (candidate, head) ----
        if (tid < 32) {
            int c = tid >> 2, hh = tid & 3, l = c + 1;
            int i = S.ctrl[0];
            float v[SS]; float mx = -1e30f;
            for (int j = 0; j < SS; ++j) {
                bool allowed = (j == 0) || ((j <= l) && (i + (j - 1) < TT));
                float s = allowed ? S.sc[hh * SS + j] : -1e30f;
                v[j] = s; mx = fmaxf(mx, s);
            }
            float sum = 0.f;
            for (int j = 0; j < SS; ++j) { v[j] = expf(v[j] - mx); sum += v[j]; }
            float inv = 1.f / sum;
            for (int j = 0; j < SS; ++j) S.attw[(c * HH + hh) * SS + j] = v[j] * inv;
        }
        __syncthreads();

        // ---- 4c. O[c] = sum_j attw * V ----
        {
            int d = tid, hh = d >> 6;
            for (int c = 0; c < LL; ++c) {
                const float* aw = S.attw + (c * HH + hh) * SS;
                float acc = 0.f;
                for (int j = 0; j < SS; ++j) acc += aw[j] * S.Vm[j * DD + d];
                S.O[c * DD + d] = acc;
            }
        }
        __syncthreads();

        // ---- 5. X0 = seq[0] + O@Wo + bo  (M=8, N=256, K=256) ----
        {
            v16h af[8];
            preload_a8_f32(S.O, DD, lane, af);
            for (int tile = w; tile < 16; tile += 8) {
                int n0 = tile << 4;
                v8f acc = {};
#pragma unroll
                for (int kk = 0; kk < 8; ++kk)
                    acc = wmma_step(af[kk], load_b_tr(woT, DD, n0, kk * 32, lane), acc);
#pragma unroll
                for (int r = 0; r < 8; ++r) {
                    int row = r + (hi << 3), col = n0 + nn;
                    if (row < LL)
                        S.X0[row * DD + col] = acc[r] + bo[col] + S.seq[col];
                }
            }
        }
        __syncthreads();

        // ---- 6. LN2 ----
        ln_row(S.X0 + w * DD, S.H2 + w * DD, ln2_g, ln2_b, lane);
        __syncthreads();

        // ---- 7. FFN1: Y = relu(H2@W1+b1) f16 (M=8, N=2048, K=256) ----
        {
            v16h af[8];
            preload_a8_f32(S.H2, DD, lane, af);
            for (int tile = w; tile < 128; tile += 8) {
                int n0 = tile << 4;
                v8f acc = {};
#pragma unroll
                for (int kk = 0; kk < 8; ++kk)
                    acc = wmma_step(af[kk], load_b_tr(w1T, DD, n0, kk * 32, lane), acc);
#pragma unroll
                for (int r = 0; r < 8; ++r) {
                    int row = r + (hi << 3), col = n0 + nn;
                    if (row < LL)
                        S.Y[row * FF + col] = (_Float16)fmaxf(acc[r] + b1f[col], 0.f);
                }
            }
        }
        __syncthreads();

        // ---- 8. FFN2: ENC = X0 + Y@W2 + b2 (M=8, N=256, K=2048), dual tile ----
        {
            int n0a = w << 4, n0b = (w + 8) << 4;
            v8f acc0 = {}, acc1 = {};
            for (int k0 = 0; k0 < FF; k0 += 32) {
                v16h a  = load_a_f16p(S.Y, FF, k0, lane);
                v16h b0 = load_b_tr(w2T, FF, n0a, k0, lane);
                v16h b1 = load_b_tr(w2T, FF, n0b, k0, lane);
                __builtin_prefetch(w2T + (size_t)(n0a + nn) * FF + k0 + 512, 0, 3);
                acc0 = wmma_step(a, b0, acc0);
                acc1 = wmma_step(a, b1, acc1);
            }
#pragma unroll
            for (int r = 0; r < 8; ++r) {
                int row = r + (hi << 3);
                if (row < LL) {
                    int ca = n0a + nn, cb = n0b + nn;
                    S.ENC[row * DD + ca] = acc0[r] + b2f[ca] + S.X0[row * DD + ca];
                    S.ENC[row * DD + cb] = acc1[r] + b2f[cb] + S.X0[row * DD + cb];
                }
            }
        }
        __syncthreads();

        // ---- 9. eoc hidden: U = relu(ENC@We1 + be1) (M=8, N=256, K=256) ----
        {
            v16h af[8];
            preload_a8_f32(S.ENC, DD, lane, af);
            for (int tile = w; tile < 16; tile += 8) {
                int n0 = tile << 4;
                v8f acc = {};
#pragma unroll
                for (int kk = 0; kk < 8; ++kk)
                    acc = wmma_step(af[kk], load_b_tr(we1T, DD, n0, kk * 32, lane), acc);
#pragma unroll
                for (int r = 0; r < 8; ++r) {
                    int row = r + (hi << 3), col = n0 + nn;
                    if (row < LL)
                        S.U[row * DD + col] = fmaxf(acc[r] + be1[col], 0.f);
                }
            }
        }
        __syncthreads();

        // ---- 10. p[c] = sigmoid(U[c].We2 + be2): one wave per candidate ----
        {
            const float* u = S.U + w * DD;
            float s = 0.f;
#pragma unroll
            for (int q = 0; q < 8; ++q) {
                int e = lane + 32 * q;
                s += u[e] * We2[e];
            }
#pragma unroll
            for (int off = 16; off; off >>= 1) s += __shfl_xor(s, off, 32);
            if (lane == 0) S.p[w] = 1.f / (1.f + expf(-(s + be2[0])));
        }
        __syncthreads();

        // ---- 11. decision (scalar) ----
        if (tid == 0) {
            int i = S.ctrl[0];
            int lmax = TT - i;
            if (lmax < 1) lmax = 1;
            if (lmax > LL) lmax = LL;
            int lstar = lmax;
            for (int c = 0; c < LL; ++c) {
                if (S.p[c] > THRESH && (c + 1) <= lmax) { lstar = c + 1; break; }
            }
            int active = (i < TT) ? 1 : 0;
            S.ctrl[2] = lstar;
            S.ctrl[3] = active;
            if (active) { S.ctrl[0] = i + lstar; S.ctrl[1] += 1; }
        }
        __syncthreads();

        // ---- 12. emit chunk row ----
        {
            int lstar = S.ctrl[2], active = S.ctrl[3];
            float v = active ? S.ENC[(lstar - 1) * DD + tid] : 0.f;
            out[((size_t)b * TT + step) * DD + tid] = v;
        }
        __syncthreads();
    }

    if (tid == 0)
        ((int*)(out + (size_t)BB * TT * DD))[b] = S.ctrl[1];  // counts (int32 bits)
}

// ---------------- host entry ----------------
extern "C" void kernel_launch(void* const* d_in, const int* in_sizes, int n_in,
                              void* d_out, int out_size, void* d_ws, size_t ws_size,
                              hipStream_t stream) {
    const float* frames = (const float*)d_in[0];
    const float* cls    = (const float*)d_in[1];
    const float* Wq = (const float*)d_in[2];  const float* bq = (const float*)d_in[3];
    const float* Wk = (const float*)d_in[4];  const float* bk = (const float*)d_in[5];
    const float* Wv = (const float*)d_in[6];  const float* bv = (const float*)d_in[7];
    const float* Wo = (const float*)d_in[8];  const float* bo = (const float*)d_in[9];
    const float* ln1_g = (const float*)d_in[10]; const float* ln1_b = (const float*)d_in[11];
    const float* ln2_g = (const float*)d_in[12]; const float* ln2_b = (const float*)d_in[13];
    const float* W1 = (const float*)d_in[14]; const float* b1f = (const float*)d_in[15];
    const float* W2 = (const float*)d_in[16]; const float* b2f = (const float*)d_in[17];
    const float* We1 = (const float*)d_in[18]; const float* be1 = (const float*)d_in[19];
    const float* We2 = (const float*)d_in[20]; const float* be2 = (const float*)d_in[21];

    _Float16* ws  = (_Float16*)d_ws;
    _Float16* wqT  = ws;             // 256x256
    _Float16* wkT  = ws + 65536;
    _Float16* wvT  = ws + 131072;
    _Float16* woT  = ws + 196608;
    _Float16* w1T  = ws + 262144;    // 2048x256
    _Float16* w2T  = ws + 786432;    // 256x2048
    _Float16* we1T = ws + 1310720;   // 256x256

    auto tr = [&](const float* s, _Float16* d, int K, int N) {
        int n = K * N;
        dc_transpose_f16<<<(n + 255) / 256, 256, 0, stream>>>(s, d, K, N);
    };
    tr(Wq, wqT, DD, DD);
    tr(Wk, wkT, DD, DD);
    tr(Wv, wvT, DD, DD);
    tr(Wo, woT, DD, DD);
    tr(W1, w1T, DD, FF);
    tr(W2, w2T, FF, DD);
    tr(We1, we1T, DD, DD);

    dim3 grid(BB), block(256);
    dc_chunker<<<grid, block, sizeof(Smem), stream>>>(
        frames, cls, bq, bk, bv, bo, ln1_g, ln1_b, ln2_g, ln2_b,
        b1f, b2f, be1, We2, be2,
        wqT, wkT, wvT, woT, w1T, w2T, we1T, (float*)d_out);
}
